// Encoder_49357764166050
// MI455X (gfx1250) — matrix operand order repaired and checked by reading, not compiled, not run on hardware
//
#include <hip/hip_runtime.h>
#include <math.h>

#define NN    8256
#define EE    132096
#define DD    32
#define HHID  128
#define NGR   32
#define PERN  258
#define OUTF  128
#define EPSV  1e-5f
#define PI_F  3.14159265358979323846f

typedef __attribute__((ext_vector_type(16))) __bf16 v16bf;
typedef __attribute__((ext_vector_type(8)))  float  v8f;

// Async global->LDS staging if this toolchain exposes the gfx1250 builtins.
#define ASYNC_LDS 0
#if defined(__has_builtin)
#if __has_builtin(__builtin_amdgcn_global_load_async_to_lds_b32) && \
    __has_builtin(__builtin_amdgcn_s_wait_asynccnt)
#undef ASYNC_LDS
#define ASYNC_LDS 1
#endif
#endif

#if ASYNC_LDS
typedef __attribute__((address_space(1))) int* gas_i32p;
typedef __attribute__((address_space(3))) int* las_i32p;
#endif

// ---------------------------------------------------------------- init
__global__ void k_init(float* __restrict__ deg, unsigned* __restrict__ maxb) {
    int i = blockIdx.x * blockDim.x + threadIdx.x;
    if (i < NN) deg[i] = 0.0f;
    if (i == NN) *maxb = 0u;
}

// ------------------------------------------- edge pre-pass: dist, max, deg
__global__ void k_edge(const float* __restrict__ pos, const int* __restrict__ ei,
                       float* __restrict__ dist, float* __restrict__ deg,
                       unsigned* __restrict__ maxb) {
    int e = blockIdx.x * blockDim.x + threadIdx.x;
    if (e >= EE) return;
    int c = ei[e], n = ei[EE + e];
    float dx = pos[c*3+0] - pos[n*3+0];
    float dy = pos[c*3+1] - pos[n*3+1];
    float dz = pos[c*3+2] - pos[n*3+2];
    float d  = sqrtf(dx*dx + dy*dy + dz*dz);
    dist[e] = d;
    atomicMax(maxb, __float_as_uint(d));   // d >= 0 -> uint order == float order
    atomicAdd(&deg[c], 1.0f);
}

// ------------------------------------------------------- node/vel encoders
__global__ void k_encode(const float* __restrict__ pos, const float* __restrict__ vel,
                         const float* __restrict__ nw1, const float* __restrict__ nb1,
                         const float* __restrict__ nw2, const float* __restrict__ nb2,
                         const float* __restrict__ vw1, const float* __restrict__ vb1,
                         const float* __restrict__ vw2, const float* __restrict__ vb2,
                         float* __restrict__ h) {
    int n = blockIdx.x * blockDim.x + threadIdx.x;
    if (n >= NN) return;
    float p0 = pos[n*3], p1 = pos[n*3+1], p2 = pos[n*3+2];
    float v0 = vel[n*3], v1 = vel[n*3+1], v2 = vel[n*3+2];
    for (int j = 0; j < 16; ++j) {
        float ap = nb2[j], av = vb2[j];
        for (int k = 0; k < HHID; ++k) {
            float tp = fmaf(p0, nw1[k], fmaf(p1, nw1[HHID+k], fmaf(p2, nw1[2*HHID+k], nb1[k])));
            tp = tp >= 0.0f ? tp : 0.2f * tp;
            ap = fmaf(tp, nw2[k*16 + j], ap);
            float tv = fmaf(v0, vw1[k], fmaf(v1, vw1[HHID+k], fmaf(v2, vw1[2*HHID+k], vb1[k])));
            tv = tv >= 0.0f ? tv : 0.2f * tv;
            av = fmaf(tv, vw2[k*16 + j], av);
        }
        h[n*DD + j]      = ap;
        h[n*DD + 16 + j] = av;
    }
}

// ----------------------------------------------- BatchNorm column statistics
__global__ void k_bnstats(const float* __restrict__ h, const float* __restrict__ g,
                          const float* __restrict__ b, float* __restrict__ scale,
                          float* __restrict__ shiftv) {
    __shared__ float ssum[256], ssq[256];
    int c = blockIdx.x, tid = threadIdx.x;
    float s = 0.0f, s2 = 0.0f;
    for (int r = tid; r < NN; r += 256) {
        float v = h[r*DD + c];
        s += v; s2 += v*v;
    }
    ssum[tid] = s; ssq[tid] = s2;
    __syncthreads();
    for (int k = 128; k > 0; k >>= 1) {
        if (tid < k) { ssum[tid] += ssum[tid+k]; ssq[tid] += ssq[tid+k]; }
        __syncthreads();
    }
    if (tid == 0) {
        float mu  = ssum[0] / (float)NN;
        float var = ssq[0] / (float)NN - mu*mu;
        float sc  = rsqrtf(var + EPSV) * g[c];
        scale[c]  = sc;
        shiftv[c] = b[c] - mu * sc;
    }
}

// -------------------------------------- hn = BN(h); zero the scatter target
__global__ void k_hn_zero(const float* __restrict__ h, const float* __restrict__ scale,
                          const float* __restrict__ shiftv, float* __restrict__ hn,
                          float* __restrict__ agg) {
    int i = blockIdx.x * blockDim.x + threadIdx.x;
    if (i >= NN*DD) return;
    int c = i & (DD-1);
    hn[i]  = fmaf(h[i], scale[c], shiftv[c]);
    agg[i] = 0.0f;
}

// ------------------------------------------------------------------ fused
// Block = 4 waves x 32 edges = 128 edges.
// Per wave: z1 -> z2 (WMMA bf16, K=128) into LDS; re-fragment z2 as two
// 16-edge A-sets. Then loop over the 64 column-tiles of em_w3: each 16-col
// slab is staged into LDS once per block (async-to-LDS when available),
// each wave builds one B-fragment set and reuses it for both A-sets, and the
// hn . We contraction is folded into per-lane register accumulators so the
// 16x1024 We tile never exists in memory.
__global__ __launch_bounds__(128)
void k_conv(const float* __restrict__ dist, const unsigned* __restrict__ maxb,
            const int* __restrict__ ei, const float* __restrict__ hn,
            const float* __restrict__ em_w1, const float* __restrict__ em_b1,
            const float* __restrict__ em_w2, const float* __restrict__ em_b2,
            const float* __restrict__ em_w3, const float* __restrict__ em_b3,
            float* __restrict__ agg) {
    __shared__ float hn_s[128][DD];          // 16 KB
    __shared__ float z2_s[128][HHID + 4];    // 67.6 KB, +4 kills bank conflicts
    __shared__ float Bs[HHID][17];           // 8.7 KB staged em_w3 slab

    const int tid  = threadIdx.x;
    const int lane = tid & 31;
    const int wv   = tid >> 5;               // wave 0..3
    const int m16  = lane & 15;              // fragment row/column id
    const int hlf  = lane >> 4;              // lane half per ISA fragment layout
    const int e0   = blockIdx.x * 128;
    const int ew0  = e0 + wv * 32;           // this wave's 32 edges

    // gather hn rows for all 128 edges (coalesced, all threads)
    for (int j = tid; j < 128 * DD; j += 128) {
        int row = j >> 5, d = j & 31;
        hn_s[row][d] = hn[ei[EE + e0 + row] * DD + d];
    }

    float maxd = __uint_as_float(*maxb);

    // ---- z2 = relu(relu(w*em_w1+b1) @ em_w2 + b2) for both 16-edge groups
    for (int g = 0; g < 2; ++g) {
        float w_m = 0.5f * (cosf(dist[ew0 + g*16 + m16] * PI_F / maxd) + 1.0f);
        v16bf a1[4];
        #pragma unroll
        for (int kb = 0; kb < 4; ++kb)
            #pragma unroll
            for (int i = 0; i < 16; ++i) {
                int k = kb*32 + hlf*16 + i;  // contiguous k order, same as B
                float z = fmaf(w_m, em_w1[k], em_b1[k]);
                a1[kb][i] = (__bf16)(z > 0.0f ? z : 0.0f);
            }
        for (int nb = 0; nb < 8; ++nb) {
            float bias = em_b2[nb*16 + m16];
            v8f c;
            #pragma unroll
            for (int r = 0; r < 8; ++r) c[r] = bias;
            #pragma unroll
            for (int kb = 0; kb < 4; ++kb) {
                v16bf b;                     // B layout: lane=col, elem i -> K=hlf*16+i
                #pragma unroll
                for (int i = 0; i < 16; ++i) {
                    int k = kb*32 + hlf*16 + i;
                    b[i] = (__bf16)em_w2[k*HHID + nb*16 + m16];
                }
                c = __builtin_amdgcn_wmma_f32_16x16x32_bf16(false, a1[kb], false, b,
                                                            (short)0, c, false, false);
            }
            #pragma unroll
            for (int r = 0; r < 8; ++r) {    // D layout: elem r -> row r + 8*hlf
                float v = c[r];
                z2_s[wv*32 + g*16 + r + 8*hlf][nb*16 + m16] = v > 0.0f ? v : 0.0f;
            }
        }
    }

    // re-fragment z2 (wave-private LDS rows; in-order within the wave)
    v16bf a2g0[4], a2g1[4];
    #pragma unroll
    for (int kb = 0; kb < 4; ++kb)
        #pragma unroll
        for (int i = 0; i < 16; ++i) {
            int k = kb*32 + hlf*16 + i;
            a2g0[kb][i] = (__bf16)z2_s[wv*32 +      m16][k];
            a2g1[kb][i] = (__bf16)z2_s[wv*32 + 16 + m16][k];
        }

    __syncthreads();                          // hn_s ready for all waves

    float msgA0[8], msgA1[8], msgB0[8], msgB1[8];
    #pragma unroll
    for (int r = 0; r < 8; ++r) { msgA0[r]=0.f; msgA1[r]=0.f; msgB0[r]=0.f; msgB1[r]=0.f; }

    // stage one 16-column em_w3 slab into LDS (shared by all 4 waves)
    auto stage = [&](int t) {
        __syncthreads();                      // everyone done reading previous slab
#if ASYNC_LDS
        for (int idx = tid; idx < HHID * 16; idx += 128) {
            int k = idx >> 4, cl = idx & 15;
            __builtin_amdgcn_global_load_async_to_lds_b32(
                (gas_i32p)(em_w3 + k*(DD*DD) + t*16 + cl),
                (las_i32p)&Bs[k][cl], 0, 0);
        }
        __builtin_amdgcn_s_wait_asynccnt(0);
#else
        for (int idx = tid; idx < HHID * 16; idx += 128) {
            int k = idx >> 4, cl = idx & 15;
            Bs[k][cl] = em_w3[k*(DD*DD) + t*16 + cl];
        }
#endif
        __syncthreads();                      // slab visible to all waves
    };

    // one We column-tile: WMMA for both groups + register-level hn contraction
    auto tile_cd = [&](int t, float* acc0, float* acc1) {
        stage(t);
        v16bf bfr[4];
        #pragma unroll
        for (int kb = 0; kb < 4; ++kb)
            #pragma unroll
            for (int i = 0; i < 16; ++i)
                bfr[kb][i] = (__bf16)Bs[kb*32 + hlf*16 + i][m16];
        const int   d    = t >> 1;            // col = t*16+n -> (d = t>>1, f = (t&1)*16+n)
        const float bias = em_b3[t*16 + m16];
        {
            v8f c;
            #pragma unroll
            for (int r = 0; r < 8; ++r) c[r] = bias;
            #pragma unroll
            for (int kb = 0; kb < 4; ++kb)
                c = __builtin_amdgcn_wmma_f32_16x16x32_bf16(false, a2g0[kb], false, bfr[kb],
                                                            (short)0, c, false, false);
            #pragma unroll
            for (int r = 0; r < 8; ++r)
                acc0[r] = fmaf(hn_s[wv*32 + r + 8*hlf][d], c[r], acc0[r]);
        }
        {
            v8f c;
            #pragma unroll
            for (int r = 0; r < 8; ++r) c[r] = bias;
            #pragma unroll
            for (int kb = 0; kb < 4; ++kb)
                c = __builtin_amdgcn_wmma_f32_16x16x32_bf16(false, a2g1[kb], false, bfr[kb],
                                                            (short)0, c, false, false);
            #pragma unroll
            for (int r = 0; r < 8; ++r)
                acc1[r] = fmaf(hn_s[wv*32 + 16 + r + 8*hlf][d], c[r], acc1[r]);
        }
    };

    for (int tp = 0; tp < 32; ++tp) {
        tile_cd(2*tp,     msgA0, msgA1);      // even tiles -> f = m16
        tile_cd(2*tp + 1, msgB0, msgB1);      // odd tiles  -> f = 16 + m16
    }

    // mean-scatter (division by degree happens in k_update)
    #pragma unroll
    for (int r = 0; r < 8; ++r) {
        int m  = r + 8*hlf;
        int c0 = ei[ew0 + m];
        atomicAdd(&agg[c0*DD +      m16], msgA0[r]);
        atomicAdd(&agg[c0*DD + 16 + m16], msgB0[r]);
        int c1 = ei[ew0 + 16 + m];
        atomicAdd(&agg[c1*DD +      m16], msgA1[r]);
        atomicAdd(&agg[c1*DD + 16 + m16], msgB1[r]);
    }
}

// ------------------------------------------- residual update h += agg/deg + b
__global__ void k_update(float* __restrict__ h, const float* __restrict__ agg,
                         const float* __restrict__ deg, const float* __restrict__ conv_b) {
    int i = blockIdx.x * blockDim.x + threadIdx.x;
    if (i >= NN*DD) return;
    int n = i >> 5, c = i & (DD-1);
    float dg = deg[n];
    float denom = dg > 1.0f ? dg : 1.0f;
    h[i] = h[i] + agg[i] / denom + conv_b[c];
}

// ---------------------------------------------------------------- LayerNorm
__global__ void k_ln(float* __restrict__ h, const float* __restrict__ g,
                     const float* __restrict__ b) {
    int n = blockIdx.x * blockDim.x + threadIdx.x;
    if (n >= NN) return;
    float v[DD]; float s = 0.0f;
    for (int c = 0; c < DD; ++c) { v[c] = h[n*DD + c]; s += v[c]; }
    float mu = s / (float)DD, s2 = 0.0f;
    for (int c = 0; c < DD; ++c) { float d = v[c] - mu; s2 += d*d; }
    float rstd = rsqrtf(s2 / (float)DD + EPSV);
    for (int c = 0; c < DD; ++c)
        h[n*DD + c] = (v[c] - mu) * rstd * g[c] + b[c];
}

// ---------------------------------------------------------------- decoder
__global__ __launch_bounds__(OUTF)
void k_decode(const float* __restrict__ h, const float* __restrict__ w1,
              const float* __restrict__ b1, const float* __restrict__ w2,
              const float* __restrict__ b2, float* __restrict__ out) {
    __shared__ float t[HHID];
    int gidx = blockIdx.x, j = threadIdx.x;
    const float* hg = h + gidx * PERN * DD;
    float acc = b1[j];
    for (int i = 0; i < PERN*DD; ++i)
        acc = fmaf(hg[i], w1[i*HHID + j], acc);
    t[j] = acc >= 0.0f ? acc : 0.2f * acc;
    __syncthreads();
    float o = b2[j];
    #pragma unroll 8
    for (int k = 0; k < HHID; ++k)
        o = fmaf(t[k], w2[k*OUTF + j], o);
    out[gidx*OUTF + j] = o;
}

// ================================================================= launcher
extern "C" void kernel_launch(void* const* d_in, const int* in_sizes, int n_in,
                              void* d_out, int out_size, void* d_ws, size_t ws_size,
                              hipStream_t stream) {
    const float* pos   = (const float*)d_in[0];
    const float* vel   = (const float*)d_in[1];
    const int*   ei    = (const int*)  d_in[2];
    const float* ne_w1 = (const float*)d_in[3];
    const float* ne_b1 = (const float*)d_in[4];
    const float* ne_w2 = (const float*)d_in[5];
    const float* ne_b2 = (const float*)d_in[6];
    const float* ve_w1 = (const float*)d_in[7];
    const float* ve_b1 = (const float*)d_in[8];
    const float* ve_w2 = (const float*)d_in[9];
    const float* ve_b2 = (const float*)d_in[10];
    const float* em_w1 = (const float*)d_in[11];
    const float* em_b1 = (const float*)d_in[12];
    const float* em_w2 = (const float*)d_in[13];
    const float* em_b2 = (const float*)d_in[14];
    const float* em_w3 = (const float*)d_in[15];
    const float* em_b3 = (const float*)d_in[16];
    const float* convb = (const float*)d_in[17];
    const float* bn_g[2] = { (const float*)d_in[18], (const float*)d_in[20] };
    const float* bn_b[2] = { (const float*)d_in[19], (const float*)d_in[21] };
    const float* ln_g  = (const float*)d_in[22];
    const float* ln_b  = (const float*)d_in[23];
    const float* fc_w1 = (const float*)d_in[24];
    const float* fc_b1 = (const float*)d_in[25];
    const float* fc_w2 = (const float*)d_in[26];
    const float* fc_b2 = (const float*)d_in[27];
    float* out = (float*)d_out;

    // workspace carve-up (~3.7 MB)
    float* wsf    = (float*)d_ws;
    float* dist   = wsf;                      // E
    float* deg    = dist + EE;                // N
    unsigned* mxb = (unsigned*)(deg + NN);    // 1
    float* h      = deg + NN + 1;             // N*D
    float* hn     = h   + NN*DD;              // N*D
    float* agg    = hn  + NN*DD;              // N*D
    float* scale  = agg + NN*DD;              // D
    float* shiftv = scale + DD;               // D

    k_init  <<<(NN + 256)/256, 256, 0, stream>>>(deg, mxb);
    k_edge  <<<(EE + 255)/256, 256, 0, stream>>>(pos, ei, dist, deg, mxb);
    k_encode<<<(NN + 127)/128, 128, 0, stream>>>(pos, vel, ne_w1, ne_b1, ne_w2, ne_b2,
                                                 ve_w1, ve_b1, ve_w2, ve_b2, h);
    for (int layer = 0; layer < 2; ++layer) {
        k_bnstats<<<DD, 256, 0, stream>>>(h, bn_g[layer], bn_b[layer], scale, shiftv);
        k_hn_zero<<<(NN*DD + 255)/256, 256, 0, stream>>>(h, scale, shiftv, hn, agg);
        k_conv   <<<EE/128, 128, 0, stream>>>(dist, mxb, ei, hn,
                                              em_w1, em_b1, em_w2, em_b2, em_w3, em_b3, agg);
        k_update <<<(NN*DD + 255)/256, 256, 0, stream>>>(h, agg, deg, convb);
    }
    k_ln    <<<(NN + 127)/128, 128, 0, stream>>>(h, ln_g, ln_b);
    k_decode<<<NGR, OUTF, 0, stream>>>(h, fc_w1, fc_b1, fc_w2, fc_b2, out);
}
